// DistanceWeightedSampling_39556648796292
// MI455X (gfx1250) — matrix operand order repaired and checked by reading, not compiled, not run on hardware
//
#include <hip/hip_runtime.h>
#include <hip/hip_bf16.h>
#include <math.h>

// ---------------------------------------------------------------------------
// DistanceWeightedSampling on MI455X (gfx1250, wave32, WMMA)
//
// v2: one-shot prep kernel splits x into bf16 hi/lo planes laid out so WMMA
// A/B fragments are contiguous loads (b128), removing all f32->bf16
// conversion VALU from the hot Gram loop. Hot loop = global_load_b128 +
// v_wmma_f32_16x16x32_bf16 (3x per K-step for f32-accurate hi/lo product)
// + streaming-softmax epilogue. Sampling via inverse-CDF over per-tile sums,
// recomputing only the crossing tile. x planes (4 MB total) are L2-resident;
// HBM traffic is dominated by the ~88 MB of gathered outputs.
// ---------------------------------------------------------------------------

typedef __attribute__((ext_vector_type(16))) __bf16 v16bf;
typedef __attribute__((ext_vector_type(8)))  __bf16 v8bf;
typedef __attribute__((ext_vector_type(8)))  float  v8f;

#define N_PTS   4096
#define DIM     256
#define KGRP    8
#define NTILES  (N_PTS / 16)          // 256 column tiles
#define RPB     16                    // anchor rows per block
#define NTRIP   (N_PTS * (KGRP - 1))  // 28672 triplets

__device__ __forceinline__ float u01hash(unsigned v) {
    v ^= v >> 16; v *= 0x7feb352dU;
    v ^= v >> 15; v *= 0x846ca68bU;
    v ^= v >> 16;
    return ((float)(v >> 8) + 0.5f) * (1.0f / 16777216.0f);
}

// log_w = (2-d)*log(dist) - (d-3)/2 * log(max(1 - dist^2/4, 1e-8)), d = 256
__device__ __forceinline__ void sim_to_lw(float sim, int gRow, int gCol,
                                          float* lw, bool* inc) {
    float sq   = 2.0f - 2.0f * sim + (gRow == gCol ? 1.0f : 0.0f); // +eye
    float dist = fmaxf(sqrtf(fmaxf(sq, 0.0f)), 0.5f);              // CUTOFF
    bool  ok   = ((gRow >> 3) != (gCol >> 3)) && (dist < 1.4f);    // mask+cut
    float v    = -254.0f * __logf(dist)
                 - 126.5f * __logf(fmaxf(1.0f - 0.25f * dist * dist, 1e-8f));
    *inc = ok;
    *lw  = ok ? v : -1e30f;
}

// ---- one-shot hi/lo bf16 split of x into L2-resident planes ---------------
__global__ __launch_bounds__(256)
void prep_bf16_kernel(const float* __restrict__ x,
                      __bf16* __restrict__ xHi, __bf16* __restrict__ xLo) {
    const size_t i = (size_t)blockIdx.x * 256 + threadIdx.x;
    float  f  = x[i];
    __bf16 hi = (__bf16)f;
    xHi[i] = hi;
    xLo[i] = (__bf16)(f - (float)hi);
}

__global__ __launch_bounds__(256)
void gram_sample_kernel(const float* __restrict__ x,
                        const __bf16* __restrict__ xHi,
                        const __bf16* __restrict__ xLo,
                        int* __restrict__ negOut) {
    __shared__ float tMax[RPB][NTILES];     // per (row,tile) log-weight max
    __shared__ float tSum[RPB][NTILES];     // per (row,tile) sum(exp(lw-max))
    __shared__ float rowM[RPB], rowS[RPB];

    const int tid   = threadIdx.x;
    const int wave  = tid >> 5;
    const int lane  = tid & 31;
    const int h     = lane >> 4;      // half-wave select
    const int nLoc  = lane & 15;
    const int rBase = blockIdx.x * RPB;

    // ---- per-wave A fragments: direct contiguous bf16 loads ---------------
    // A 16x32 bf16 layout: lane&15 = M; K chunk [32s+16h,+16)
    const __bf16* aRowHi = xHi + (size_t)(rBase + nLoc) * DIM;
    const __bf16* aRowLo = xLo + (size_t)(rBase + nLoc) * DIM;
    v16bf aHi[8], aLo[8];
    #pragma unroll
    for (int s = 0; s < 8; ++s) {
        const int k0 = 32 * s + h * 16;
        v8bf h0 = *(const v8bf*)(aRowHi + k0);
        v8bf h1 = *(const v8bf*)(aRowHi + k0 + 8);
        v8bf l0 = *(const v8bf*)(aRowLo + k0);
        v8bf l1 = *(const v8bf*)(aRowLo + k0 + 8);
        aHi[s] = __builtin_shufflevector(h0, h1, 0, 1, 2, 3, 4, 5, 6, 7,
                                         8, 9, 10, 11, 12, 13, 14, 15);
        aLo[s] = __builtin_shufflevector(l0, l1, 0, 1, 2, 3, 4, 5, 6, 7,
                                         8, 9, 10, 11, 12, 13, 14, 15);
    }

    // ---- sweep column tiles: WMMA + streaming softmax stats ---------------
    for (int t = wave; t < NTILES; t += 8) {
        const int colBase = t * 16;
        const int col     = colBase + nLoc;        // B: lane&15 = N (= x row)
        const __bf16* bColHi = xHi + (size_t)col * DIM;
        const __bf16* bColLo = xLo + (size_t)col * DIM;
        v8f acc = {0.f, 0.f, 0.f, 0.f, 0.f, 0.f, 0.f, 0.f};
        #pragma unroll
        for (int s = 0; s < 8; ++s) {
            const int kb = 32 * s + h * 16;        // 16 consecutive K per lane
            v16bf bHi = *(const v16bf*)(bColHi + kb);
            v16bf bLo = *(const v16bf*)(bColLo + kb);
            // f32-accurate: hiA*hiB + hiA*loB + loA*hiB
            acc = __builtin_amdgcn_wmma_f32_16x16x32_bf16(
                false, aHi[s], false, bHi, (short)0, acc, false, false);
            acc = __builtin_amdgcn_wmma_f32_16x16x32_bf16(
                false, aHi[s], false, bLo, (short)0, acc, false, false);
            acc = __builtin_amdgcn_wmma_f32_16x16x32_bf16(
                false, aLo[s], false, bHi, (short)0, acc, false, false);
        }
        // D layout: lane&15 = N, VGPR v -> M = v + 8*h
        #pragma unroll
        for (int v = 0; v < 8; ++v) {
            int   rowLoc = v + 8 * h;
            float lw; bool inc;
            sim_to_lw(acc[v], rBase + rowLoc, colBase + nLoc, &lw, &inc);
            float mx = lw;
            #pragma unroll
            for (int o = 1; o < 16; o <<= 1) mx = fmaxf(mx, __shfl_xor(mx, o));
            float e  = (lw > -1e29f) ? __expf(lw - mx) : 0.0f;
            float sm = e;
            #pragma unroll
            for (int o = 1; o < 16; o <<= 1) sm += __shfl_xor(sm, o);
            if (nLoc == 0) { tMax[rowLoc][t] = mx; tSum[rowLoc][t] = sm; }
        }
    }
    __syncthreads();

    // ---- per-row combine: global max, rescale tile sums into a CDF base ---
    if (tid < RPB) {
        const int r = tid;
        float M = -1e30f;
        for (int j = 0; j < NTILES; ++j) M = fmaxf(M, tMax[r][j]);
        float S = 0.0f;
        for (int j = 0; j < NTILES; ++j) {
            float tm = tMax[r][j];
            float vs = (tm > -1e29f) ? tSum[r][j] * __expf(tm - M) : 0.0f;
            tSum[r][j] = vs;
            S += vs;
        }
        rowM[r] = M; rowS[r] = S;
    }
    __syncthreads();

    // ---- inverse-CDF sampling of 7 negatives per row (one wave per pair) --
    for (int p = wave; p < RPB * (KGRP - 1); p += 8) {
        const int r    = p / (KGRP - 1);
        const int tIdx = p % (KGRP - 1);
        const int gRow = rBase + r;
        const float u  = u01hash((unsigned)(gRow * 7 + tIdx) * 2654435761u
                                 + 0x9E3779B9u);
        const float S  = rowS[r];
        int neg;
        if (!(S > 0.0f)) {
            // wsum==0 -> uniform over columns outside own block of 8
            int idx = (int)(u * (float)(N_PTS - KGRP));
            if (idx > N_PTS - KGRP - 1) idx = N_PTS - KGRP - 1;
            int ob = (gRow >> 3) << 3;
            neg = idx + (idx >= ob ? KGRP : 0);
        } else {
            const float target = u * S;
            int jt = -1; float cumB = 0.0f;
            if (lane == 0) {
                float c = 0.0f;
                for (int j = 0; j < NTILES; ++j) {
                    float nv = c + tSum[r][j];
                    if (nv >= target) { jt = j; cumB = c; break; }
                    c = nv;
                }
                if (jt < 0) { jt = NTILES - 1; cumB = c - tSum[r][NTILES - 1]; }
            }
            jt   = __shfl(jt, 0);
            cumB = __shfl(cumB, 0);
            const float tloc = target - cumB;
            float w = 0.0f;
            const int colG = jt * 16 + nLoc;
            if (lane < 16) {                       // recompute just this tile
                const float* ar = x + (size_t)gRow * DIM;
                const float* br = x + (size_t)colG * DIM;
                float dot = 0.0f;
                for (int kk = 0; kk < DIM; ++kk)
                    dot = fmaf(ar[kk], br[kk], dot);
                float lw; bool inc;
                sim_to_lw(dot, gRow, colG, &lw, &inc);
                w = inc ? __expf(lw - rowM[r]) : 0.0f;
            }
            int pick = -1, lastInc = -1;
            float c2 = 0.0f;
            for (int jj = 0; jj < 16; ++jj) {
                float wj = __shfl(w, jj);
                if (wj > 0.0f) lastInc = jt * 16 + jj;
                c2 += wj;
                if (pick < 0 && wj > 0.0f && c2 >= tloc) pick = jt * 16 + jj;
            }
            if (pick < 0) pick = lastInc;
            if (pick < 0) {                         // numeric safety net
                int idx = (int)(u * (float)(N_PTS - KGRP));
                if (idx > N_PTS - KGRP - 1) idx = N_PTS - KGRP - 1;
                int ob = (gRow >> 3) << 3;
                pick = idx + (idx >= ob ? KGRP : 0);
            }
            neg = pick;
        }
        if (lane == 0) negOut[gRow * (KGRP - 1) + tIdx] = neg;
    }
}

// ---------------------------------------------------------------------------
// Triplet assembly + bandwidth-bound gathers (a,p analytic, n from scratch).
// Output layout: [triplets (T*3)] [x[a] (T*256)] [x[p]] [x[n]]  as f32.
// ---------------------------------------------------------------------------
__global__ __launch_bounds__(256)
void gather_kernel(const float* __restrict__ x, const int* __restrict__ neg,
                   float* __restrict__ out) {
    const int t   = blockIdx.x;            // 0..NTRIP-1
    const int a   = t / (KGRP - 1);
    const int j   = t % (KGRP - 1);
    const int m   = a & (KGRP - 1);
    const int blk = a >> 3;
    const int pos = (blk << 3) + j + (j >= m ? 1 : 0);
    const int nn  = neg[t] & (N_PTS - 1);

    const size_t T  = (size_t)NTRIP;
    float* trip = out;
    float* xa   = out + 3 * T;
    float* xp   = xa + T * DIM;
    float* xn   = xp + T * DIM;

    const int k = threadIdx.x;             // 256 threads = one row element each
    xa[(size_t)t * DIM + k] = x[(size_t)a   * DIM + k];
    xp[(size_t)t * DIM + k] = x[(size_t)pos * DIM + k];
    xn[(size_t)t * DIM + k] = x[(size_t)nn  * DIM + k];
    if (k == 0) {
        trip[(size_t)t * 3 + 0] = (float)a;
        trip[(size_t)t * 3 + 1] = (float)pos;
        trip[(size_t)t * 3 + 2] = (float)nn;
    }
}

extern "C" void kernel_launch(void* const* d_in, const int* in_sizes, int n_in,
                              void* d_out, int out_size, void* d_ws, size_t ws_size,
                              hipStream_t stream) {
    (void)in_sizes; (void)n_in; (void)out_size; (void)ws_size;
    const float* x = (const float*)d_in[0];
    float*       out = (float*)d_out;

    // d_ws layout: xHi bf16[4096*256] | xLo bf16[4096*256] | neg int[28672]
    __bf16* xHi = (__bf16*)d_ws;
    __bf16* xLo = xHi + (size_t)N_PTS * DIM;
    int*    nws = (int*)(xLo + (size_t)N_PTS * DIM);

    prep_bf16_kernel<<<(N_PTS * DIM) / 256, 256, 0, stream>>>(x, xHi, xLo);
    gram_sample_kernel<<<N_PTS / RPB, 256, 0, stream>>>(x, xHi, xLo, nws);
    gather_kernel<<<NTRIP, 256, 0, stream>>>(x, nws, out);
}
